// AttnDecoderRNN_31069793419640
// MI455X (gfx1250) — matrix operand order repaired and checked by reading, not compiled, not run on hardware
//
#include <hip/hip_runtime.h>
#include <hip/hip_bf16.h>
#include <math.h>

// Problem constants (from reference)
static constexpr int Bn = 64;      // batch
static constexpr int Sn = 64;      // source length
static constexpr int Tn = 15;      // decode steps
static constexpr int Hn = 1024;    // hidden
static constexpr int Vn = 50257;   // vocab

typedef __attribute__((ext_vector_type(16))) _Float16 v16h;
typedef __attribute__((ext_vector_type(8)))  _Float16 v8h;
typedef __attribute__((ext_vector_type(8)))  float    v8f;
typedef __attribute__((ext_vector_type(4)))  float    v4f;

// ---------------------------------------------------------------------------
// WMMA fragment loaders (wave32; contiguous k order, same for A and B)
// A 16x32 f16: half=0 holds K {k0..k0+15}; half=1 offset +16
// B 32x16 f16: lane n holds col n; half=0 -> K {k0..k0+15}, half=1 -> +16
// ---------------------------------------------------------------------------
__device__ inline v16h ld_a16(const _Float16* __restrict__ p) {
    // p = A16[row] + k0 + half*16 ; two 16B runs at +0 and +8 halves
    v8h a = *(const v8h*)(p + 0);
    v8h b = *(const v8h*)(p + 8);
    v16h r;
#pragma unroll
    for (int j = 0; j < 8; ++j) { r[j] = a[j]; r[8 + j] = b[j]; }
    return r;
}

__device__ inline v16h ld_b16(const _Float16* __restrict__ p) {
    return *(const v16h*)p;    // 32B contiguous -> 2x b128
}

__device__ inline v16h ld_a32(const float* __restrict__ p) {
    v4f a = *(const v4f*)(p + 0);
    v4f b = *(const v4f*)(p + 4);
    v4f c = *(const v4f*)(p + 8);
    v4f d = *(const v4f*)(p + 12);
    v16h r;
#pragma unroll
    for (int j = 0; j < 4; ++j) {
        r[j]      = (_Float16)a[j];
        r[4 + j]  = (_Float16)b[j];
        r[8 + j]  = (_Float16)c[j];
        r[12 + j] = (_Float16)d[j];
    }
    return r;
}

__device__ inline v16h ld_b32(const float* __restrict__ p) {
    v4f a = *(const v4f*)(p + 0);
    v4f b = *(const v4f*)(p + 4);
    v4f c = *(const v4f*)(p + 8);
    v4f d = *(const v4f*)(p + 12);
    v16h r;
#pragma unroll
    for (int j = 0; j < 4; ++j) {
        r[j]      = (_Float16)a[j];
        r[4 + j]  = (_Float16)b[j];
        r[8 + j]  = (_Float16)c[j];
        r[12 + j] = (_Float16)d[j];
    }
    return r;
}

// ---------------------------------------------------------------------------
// Primary GEMM (all-f16 operands): C[M,N] = A16[M,K] @ W16[N,K]^T + bias
// One wave computes a 64x16 C tile; B fragment reused across 4 m-subtiles.
// Unroll-by-2 ping-pong pipeline: fragments for chunk k+32 are loaded while
// chunk k's WMMAs issue; buffers are reloaded in place (no register copies).
// Requires M % 64 == 0, K % 64 == 0.
// ---------------------------------------------------------------------------
__global__ __launch_bounds__(256) void gemm_wmma_h(
    const _Float16* __restrict__ A, const _Float16* __restrict__ W,
    const float* __restrict__ bias, float* __restrict__ C,
    int M, int N, int K, int ldc)
{
    int wave = (blockIdx.x * blockDim.x + threadIdx.x) >> 5;
    int lane = threadIdx.x & 31;
    int tilesN = (N + 15) >> 4;
    int tilesM = M >> 6;
    if (wave >= tilesM * tilesN) return;
    int tm = wave / tilesN;
    int tn = wave % tilesN;
    int m0 = tm << 6;
    int n0 = tn << 4;
    int hv = lane >> 4;
    int lr = lane & 15;

    int ncol = n0 + lr;
    int nrow = (ncol < N) ? ncol : (N - 1);
    const _Float16* wb = W + (size_t)nrow * K + hv * 16;
    const _Float16* ab = A + (size_t)(m0 + lr) * K + hv * 16;
    const size_t arow = (size_t)16 * K;   // per-m-subtile stride

    v8f acc[4] = {};

    // prologue: chunk 0 into buffer 0
    v16h b0 = ld_b16(wb);
    v16h a0[4];
#pragma unroll
    for (int mt = 0; mt < 4; ++mt) a0[mt] = ld_a16(ab + mt * arow);

    v16h b1;
    v16h a1[4];

    int k0 = 0;
    for (; k0 < K - 64; k0 += 64) {
        // load chunk k0+32 into buffer 1
        b1 = ld_b16(wb + k0 + 32);
#pragma unroll
        for (int mt = 0; mt < 4; ++mt) a1[mt] = ld_a16(ab + mt * arow + k0 + 32);
        __builtin_prefetch((const void*)(wb + k0 + 128), 0, 0);
        // compute chunk k0 (buffer 0)
#pragma unroll
        for (int mt = 0; mt < 4; ++mt)
            acc[mt] = __builtin_amdgcn_wmma_f32_16x16x32_f16(
                false, a0[mt], false, b0, (short)0, acc[mt], false, false);
        // load chunk k0+64 into buffer 0
        b0 = ld_b16(wb + k0 + 64);
#pragma unroll
        for (int mt = 0; mt < 4; ++mt) a0[mt] = ld_a16(ab + mt * arow + k0 + 64);
        // compute chunk k0+32 (buffer 1)
#pragma unroll
        for (int mt = 0; mt < 4; ++mt)
            acc[mt] = __builtin_amdgcn_wmma_f32_16x16x32_f16(
                false, a1[mt], false, b1, (short)0, acc[mt], false, false);
    }
    // epilogue: chunks k0 (in buffer 0) and k0+32
    b1 = ld_b16(wb + k0 + 32);
#pragma unroll
    for (int mt = 0; mt < 4; ++mt) a1[mt] = ld_a16(ab + mt * arow + k0 + 32);
#pragma unroll
    for (int mt = 0; mt < 4; ++mt)
        acc[mt] = __builtin_amdgcn_wmma_f32_16x16x32_f16(
            false, a0[mt], false, b0, (short)0, acc[mt], false, false);
#pragma unroll
    for (int mt = 0; mt < 4; ++mt)
        acc[mt] = __builtin_amdgcn_wmma_f32_16x16x32_f16(
            false, a1[mt], false, b1, (short)0, acc[mt], false, false);

    if (ncol < N) {
        float bb = bias ? bias[nrow] : 0.0f;
#pragma unroll
        for (int mt = 0; mt < 4; ++mt)
#pragma unroll
            for (int j = 0; j < 8; ++j)
                C[(size_t)(m0 + mt * 16 + hv * 8 + j) * ldc + ncol] = acc[mt][j] + bb;
    }
}

// ---------------------------------------------------------------------------
// Fallback GEMM (f32 operands, in-loop conversion) — used if ws too small
// ---------------------------------------------------------------------------
__global__ __launch_bounds__(256) void gemm_wmma_f(
    const float* __restrict__ A, const float* __restrict__ W,
    const float* __restrict__ bias, float* __restrict__ C,
    int M, int N, int K, int ldc)
{
    int wave = (blockIdx.x * blockDim.x + threadIdx.x) >> 5;
    int lane = threadIdx.x & 31;
    int tilesN = (N + 15) >> 4;
    int tilesM = M >> 6;
    if (wave >= tilesM * tilesN) return;
    int tm = wave / tilesN;
    int tn = wave % tilesN;
    int m0 = tm << 6;
    int n0 = tn << 4;
    int hv = lane >> 4;
    int lr = lane & 15;

    int ncol = n0 + lr;
    int nrow = (ncol < N) ? ncol : (N - 1);
    const float* wb = W + (size_t)nrow * K + hv * 16;
    const float* ab = A + (size_t)(m0 + lr) * K + hv * 16;

    v8f acc[4] = {};
    for (int k0 = 0; k0 < K; k0 += 32) {
        v16h bf = ld_b32(wb + k0);
#pragma unroll
        for (int mt = 0; mt < 4; ++mt) {
            v16h af = ld_a32(ab + (size_t)mt * 16 * K + k0);
            acc[mt] = __builtin_amdgcn_wmma_f32_16x16x32_f16(
                false, af, false, bf, (short)0, acc[mt], false, false);
        }
    }
    if (ncol < N) {
        float bb = bias ? bias[nrow] : 0.0f;
#pragma unroll
        for (int mt = 0; mt < 4; ++mt)
#pragma unroll
            for (int j = 0; j < 8; ++j)
                C[(size_t)(m0 + mt * 16 + hv * 8 + j) * ldc + ncol] = acc[mt][j] + bb;
    }
}

static inline void launch_gemm_h(const _Float16* A, const _Float16* W,
                                 const float* bias, float* C,
                                 int M, int N, int K, int ldc, hipStream_t s) {
    int tiles = (M >> 6) * ((N + 15) >> 4);
    gemm_wmma_h<<<(tiles + 7) / 8, 256, 0, s>>>(A, W, bias, C, M, N, K, ldc);
}
static inline void launch_gemm_f(const float* A, const float* W,
                                 const float* bias, float* C,
                                 int M, int N, int K, int ldc, hipStream_t s) {
    int tiles = (M >> 6) * ((N + 15) >> 4);
    gemm_wmma_f<<<(tiles + 7) / 8, 256, 0, s>>>(A, W, bias, C, M, N, K, ldc);
}

// ---------------------------------------------------------------------------
// f32 -> f16 bulk conversion (one-time for weights / encoder outputs)
// ---------------------------------------------------------------------------
__global__ __launch_bounds__(256) void cvt_f16_kernel(
    const float* __restrict__ src, _Float16* __restrict__ dst, size_t n) {
    size_t i = ((size_t)blockIdx.x * blockDim.x + threadIdx.x) * 4;
    if (i + 3 < n) {
        v4f v = *(const v4f*)(src + i);
        _Float16 o0 = (_Float16)v[0], o1 = (_Float16)v[1];
        _Float16 o2 = (_Float16)v[2], o3 = (_Float16)v[3];
        dst[i] = o0; dst[i + 1] = o1; dst[i + 2] = o2; dst[i + 3] = o3;
    } else {
        for (; i < n; ++i) dst[i] = (_Float16)src[i];
    }
}

// ---------------------------------------------------------------------------
// Embedding gather: x[b, 0:H] = emb[tok_t[b]]  (templated output precision)
// ---------------------------------------------------------------------------
template <typename CT>
__global__ __launch_bounds__(256) void embed_kernel(
    const float* __restrict__ emb, const int* __restrict__ target,
    CT* __restrict__ x, int t) {
    int b = blockIdx.x;
    int tok = (t == 0) ? 0 : target[b * Tn + (t - 1)];
    const float* e = emb + (size_t)tok * Hn;
    for (int i = threadIdx.x; i < Hn; i += 256)
        x[(size_t)b * (2 * Hn) + i] = (CT)e[i];
}

// ---------------------------------------------------------------------------
// Attention: scores -> softmax -> attn weights (d_out) + ctx (x[:, H:2H])
// ---------------------------------------------------------------------------
template <typename CT>
__global__ __launch_bounds__(256) void attn_kernel(
    const float* __restrict__ q,      // [B,H]
    const float* __restrict__ Uk,     // [B,S,H]
    const float* __restrict__ Va,     // [H]
    const float* __restrict__ bvv,    // [1]
    const float* __restrict__ enc,    // [B,S,H]
    float* __restrict__ attn_out,     // base = att + t*S ; index b*T*S + s
    CT* __restrict__ ctx_out)         // base = x + H ; index b*2H + i
{
    __shared__ float sw[Sn];
    int b = blockIdx.x;
    int wave = threadIdx.x >> 5;
    int lane = threadIdx.x & 31;

    const float* qq = q + (size_t)b * Hn;
    for (int s = wave; s < Sn; s += 8) {
        const float* uk = Uk + ((size_t)b * Sn + s) * Hn;
        float acc = 0.0f;
        for (int h = lane; h < Hn; h += 32)
            acc += tanhf(qq[h] + uk[h]) * Va[h];
#pragma unroll
        for (int off = 16; off > 0; off >>= 1)
            acc += __shfl_xor(acc, off, 32);
        if (lane == 0) sw[s] = acc + bvv[0];
    }
    __syncthreads();

    if (wave == 0) {
        float s0 = sw[lane], s1 = sw[lane + 32];
        float m = fmaxf(s0, s1);
#pragma unroll
        for (int off = 16; off > 0; off >>= 1)
            m = fmaxf(m, __shfl_xor(m, off, 32));
        float e0 = expf(s0 - m), e1 = expf(s1 - m);
        float sum = e0 + e1;
#pragma unroll
        for (int off = 16; off > 0; off >>= 1)
            sum += __shfl_xor(sum, off, 32);
        float inv = 1.0f / sum;
        sw[lane] = e0 * inv;
        sw[lane + 32] = e1 * inv;
    }
    __syncthreads();

    if (threadIdx.x < Sn)
        attn_out[(size_t)b * (Tn * Sn) + threadIdx.x] = sw[threadIdx.x];

    const float* eb = enc + (size_t)b * Sn * Hn;
    for (int h = threadIdx.x; h < Hn; h += 256) {
        float acc = 0.0f;
#pragma unroll 4
        for (int s = 0; s < Sn; ++s)
            acc += sw[s] * eb[(size_t)s * Hn + h];
        ctx_out[(size_t)b * (2 * Hn) + h] = (CT)acc;
    }
}

// ---------------------------------------------------------------------------
// GRU update; also emits f16 copy of h_new (for WMMA) and optional hT output
// ---------------------------------------------------------------------------
__global__ __launch_bounds__(256) void gru_kernel(
    const float* __restrict__ gi, const float* __restrict__ gh,
    const float* __restrict__ h, float* __restrict__ hnew,
    _Float16* __restrict__ hnew16, float* __restrict__ hT_out) {
    int b = blockIdx.x;
    const float* gib = gi + (size_t)b * 3 * Hn;
    const float* ghb = gh + (size_t)b * 3 * Hn;
    for (int i = threadIdx.x; i < Hn; i += 256) {
        float ir = gib[i],          hr = ghb[i];
        float iz = gib[Hn + i],     hz = ghb[Hn + i];
        float in = gib[2 * Hn + i], hn = ghb[2 * Hn + i];
        float r = 1.0f / (1.0f + expf(-(ir + hr)));
        float z = 1.0f / (1.0f + expf(-(iz + hz)));
        float n = tanhf(in + r * hn);
        float o = (1.0f - z) * n + z * h[(size_t)b * Hn + i];
        hnew[(size_t)b * Hn + i] = o;
        if (hnew16) hnew16[(size_t)b * Hn + i] = (_Float16)o;
        if (hT_out) hT_out[(size_t)b * Hn + i] = o;
    }
}

// ---------------------------------------------------------------------------
// In-place log-softmax over one vocab row per block
// ---------------------------------------------------------------------------
__global__ __launch_bounds__(256) void logsoftmax_kernel(
    float* __restrict__ base, int ldr) {
    __shared__ float redm[8], reds[8];
    int b = blockIdx.x;
    float* row = base + (size_t)b * ldr;
    int lane = threadIdx.x & 31, wave = threadIdx.x >> 5;

    float m = -3.402823466e38f;
    for (int i = threadIdx.x; i < Vn; i += 256) m = fmaxf(m, row[i]);
#pragma unroll
    for (int off = 16; off > 0; off >>= 1) m = fmaxf(m, __shfl_xor(m, off, 32));
    if (lane == 0) redm[wave] = m;
    __syncthreads();
    float bm = redm[0];
#pragma unroll
    for (int i = 1; i < 8; ++i) bm = fmaxf(bm, redm[i]);

    float sum = 0.0f;
    for (int i = threadIdx.x; i < Vn; i += 256) sum += expf(row[i] - bm);
#pragma unroll
    for (int off = 16; off > 0; off >>= 1) sum += __shfl_xor(sum, off, 32);
    if (lane == 0) reds[wave] = sum;
    __syncthreads();
    float tot = 0.0f;
#pragma unroll
    for (int i = 0; i < 8; ++i) tot += reds[i];
    float lse = bm + logf(tot);

    for (int i = threadIdx.x; i < Vn; i += 256) row[i] -= lse;
}

static inline void launch_cvt(const float* s, _Float16* d, size_t n, hipStream_t st) {
    size_t q = (n + 3) / 4;
    cvt_f16_kernel<<<(unsigned)((q + 255) / 256), 256, 0, st>>>(s, d, n);
}

// ---------------------------------------------------------------------------
// Host orchestration
// ---------------------------------------------------------------------------
extern "C" void kernel_launch(void* const* d_in, const int* in_sizes, int n_in,
                              void* d_out, int out_size, void* d_ws, size_t ws_size,
                              hipStream_t stream) {
    (void)in_sizes; (void)n_in; (void)out_size;
    const float* enc    = (const float*)d_in[0];
    const float* h0src  = (const float*)d_in[1];
    const int*   target = (const int*)  d_in[2];
    const float* emb    = (const float*)d_in[3];
    const float* Wa     = (const float*)d_in[4];
    const float* ba     = (const float*)d_in[5];
    const float* Ua     = (const float*)d_in[6];
    const float* bu     = (const float*)d_in[7];
    const float* Va     = (const float*)d_in[8];
    const float* bvv    = (const float*)d_in[9];
    const float* W_ih   = (const float*)d_in[10];
    const float* W_hh   = (const float*)d_in[11];
    const float* b_ih   = (const float*)d_in[12];
    const float* b_hh   = (const float*)d_in[13];
    const float* Wout   = (const float*)d_in[14];
    const float* bout   = (const float*)d_in[15];
    float* out = (float*)d_out;

    // Output layout: log_probs [B,T,V], hT [1,B,H], attn [B,T,S]
    const size_t OUT_HT  = (size_t)Bn * Tn * Vn;
    const size_t OUT_ATT = OUT_HT + (size_t)Bn * Hn;

    // Workspace layout — always-needed f32 region first
    char* wsb = (char*)d_ws;
    size_t o = 0;
    float* UkB = (float*)(wsb + o); o += (size_t)Bn * Sn * Hn * 4;
    float* qB  = (float*)(wsb + o); o += (size_t)Bn * Hn * 4;
    float* giB = (float*)(wsb + o); o += (size_t)Bn * 3 * Hn * 4;
    float* ghB = (float*)(wsb + o); o += (size_t)Bn * 3 * Hn * 4;
    float* hB0 = (float*)(wsb + o); o += (size_t)Bn * Hn * 4;
    float* hB1 = (float*)(wsb + o); o += (size_t)Bn * Hn * 4;
    float* xBf = (float*)(wsb + o); o += (size_t)Bn * 2 * Hn * 4;  // fallback x
    // f16 region
    _Float16* x16   = (_Float16*)(wsb + o); o += (size_t)Bn * 2 * Hn * 2;
    _Float16* h16a  = (_Float16*)(wsb + o); o += (size_t)Bn * Hn * 2;
    _Float16* h16b  = (_Float16*)(wsb + o); o += (size_t)Bn * Hn * 2;
    _Float16* enc16 = (_Float16*)(wsb + o); o += (size_t)Bn * Sn * Hn * 2;
    _Float16* Wa16  = (_Float16*)(wsb + o); o += (size_t)Hn * Hn * 2;
    _Float16* Ua16  = (_Float16*)(wsb + o); o += (size_t)Hn * Hn * 2;
    _Float16* Wih16 = (_Float16*)(wsb + o); o += (size_t)3 * Hn * 2 * Hn * 2;
    _Float16* Whh16 = (_Float16*)(wsb + o); o += (size_t)3 * Hn * Hn * 2;
    _Float16* Wo16  = (_Float16*)(wsb + o); o += (size_t)Vn * Hn * 2;
    const bool f16path = (ws_size >= o);

    float* hbuf[2] = {hB0, hB1};
    _Float16* h16buf[2] = {h16a, h16b};

    hipMemcpyAsync(hB0, h0src, (size_t)Bn * Hn * 4, hipMemcpyDeviceToDevice, stream);

    if (f16path) {
        launch_cvt(h0src, h16a, (size_t)Bn * Hn, stream);
        launch_cvt(enc,   enc16, (size_t)Bn * Sn * Hn, stream);
        launch_cvt(Wa,    Wa16,  (size_t)Hn * Hn, stream);
        launch_cvt(Ua,    Ua16,  (size_t)Hn * Hn, stream);
        launch_cvt(W_ih,  Wih16, (size_t)3 * Hn * 2 * Hn, stream);
        launch_cvt(W_hh,  Whh16, (size_t)3 * Hn * Hn, stream);
        launch_cvt(Wout,  Wo16,  (size_t)Vn * Hn, stream);
        // Uk = enc @ Ua^T + bu
        launch_gemm_h(enc16, Ua16, bu, UkB, Bn * Sn, Hn, Hn, Hn, stream);
    } else {
        launch_gemm_f(enc, Ua, bu, UkB, Bn * Sn, Hn, Hn, Hn, stream);
    }

    for (int t = 0; t < Tn; ++t) {
        const float* h = hbuf[t & 1];
        float* hn = hbuf[(t + 1) & 1];
        _Float16* h16  = h16buf[t & 1];
        _Float16* hn16 = h16buf[(t + 1) & 1];

        if (f16path) {
            launch_gemm_h(h16, Wa16, ba, qB, Bn, Hn, Hn, Hn, stream);
            attn_kernel<_Float16><<<Bn, 256, 0, stream>>>(
                qB, UkB, Va, bvv, enc, out + OUT_ATT + (size_t)t * Sn, x16 + Hn);
            embed_kernel<_Float16><<<Bn, 256, 0, stream>>>(emb, target, x16, t);
            launch_gemm_h(x16, Wih16, b_ih, giB, Bn, 3 * Hn, 2 * Hn, 3 * Hn, stream);
            launch_gemm_h(h16, Whh16, b_hh, ghB, Bn, 3 * Hn, Hn, 3 * Hn, stream);
            gru_kernel<<<Bn, 256, 0, stream>>>(giB, ghB, h, hn, hn16,
                (t == Tn - 1) ? (out + OUT_HT) : nullptr);
            launch_gemm_h(hn16, Wo16, bout, out + (size_t)t * Vn,
                          Bn, Vn, Hn, Tn * Vn, stream);
        } else {
            launch_gemm_f(h, Wa, ba, qB, Bn, Hn, Hn, Hn, stream);
            attn_kernel<float><<<Bn, 256, 0, stream>>>(
                qB, UkB, Va, bvv, enc, out + OUT_ATT + (size_t)t * Sn, xBf + Hn);
            embed_kernel<float><<<Bn, 256, 0, stream>>>(emb, target, xBf, t);
            launch_gemm_f(xBf, W_ih, b_ih, giB, Bn, 3 * Hn, 2 * Hn, 3 * Hn, stream);
            launch_gemm_f(h, W_hh, b_hh, ghB, Bn, 3 * Hn, Hn, 3 * Hn, stream);
            gru_kernel<<<Bn, 256, 0, stream>>>(giB, ghB, h, hn, nullptr,
                (t == Tn - 1) ? (out + OUT_HT) : nullptr);
            launch_gemm_f(hn, Wout, bout, out + (size_t)t * Vn,
                          Bn, Vn, Hn, Tn * Vn, stream);
        }
        logsoftmax_kernel<<<Bn, 256, 0, stream>>>(out + (size_t)t * Vn, Tn * Vn);
    }
}
